// MultiHeadSelfAttention_841813590315
// MI455X (gfx1250) — compile-verified, matrix operand order checked by reading, not run on hardware
//
#include <hip/hip_runtime.h>
#include <hip/hip_bf16.h>

// ---------------------------------------------------------------------------
// MI455X (gfx1250, CDNA5, wave32) fused multi-head self-attention.
// All GEMM stages use v_wmma_f32_16x16x32_f16 (f16 in, f32 accumulate).
// Flash stage stages K/V blocks into LDS with the Tensor Data Mover
// (tensor_load_to_lds + s_wait_tensorcnt), double-buffered, shared by the
// 8 waves of a workgroup (8x L2 traffic reduction vs per-wave fetch).
// ---------------------------------------------------------------------------

typedef __attribute__((ext_vector_type(16))) _Float16 v16h;
typedef __attribute__((ext_vector_type(8)))  _Float16 v8h;
typedef __attribute__((ext_vector_type(8)))  float    v8f;
typedef __attribute__((ext_vector_type(4)))  uint32_t u32x4;
typedef __attribute__((ext_vector_type(8)))  uint32_t u32x8;

// Problem constants (from reference: B=4, N=2048, D=128, H=8)
#define BATCH 4
#define SEQ   2048
#define HDIM  128
#define NHEAD 8
#define MODEL 1024              // H*D
#define ROWS  (BATCH * SEQ)     // 8192

__device__ inline v8f wmma_f16(v16h a, v16h b, v8f c) {
  return __builtin_amdgcn_wmma_f32_16x16x32_f16(
      /*neg_a=*/false, a, /*neg_b=*/false, b,
      /*c_mod=*/(short)0, c, /*reuse_a=*/false, /*reuse_b=*/false);
}

// Load a 16x32 f16 fragment in CDNA5 A-layout (also used for B fragments:
// "row" = output column, elements = K) from a row-major matrix.
// Per ISA 7.12.2: lanes 0-15 hold rows 0-15 with K 0-7 / 16-23;
// lanes 16-31 hold the same rows with K 8-15 / 24-31.
__device__ inline v16h load_frag16x32(const _Float16* base, int ld) {
  const int lane = threadIdx.x & 31;
  const int m  = lane & 15;
  const int kb = (lane >> 4) << 3;          // 0 or 8
  const _Float16* r = base + (size_t)m * ld + kb;
  v8h lo = *(const v8h*)(r);                // K = kb .. kb+7
  v8h hi = *(const v8h*)(r + 16);           // K = 16+kb .. 16+kb+7
  v16h out;
#pragma unroll
  for (int i = 0; i < 8; ++i) { out[i] = lo[i]; out[8 + i] = hi[i]; }
  return out;
}

// Generic pointer to LDS byte offset: for LDS apertures the low 32 bits of
// the generic address are the LDS byte address (ISA 10.2 aperture mapping).
__device__ inline uint32_t lds_offset(const void* p) {
  return (uint32_t)(uintptr_t)p;
}

// ---------------------------------------------------------------------------
// Tensor Data Mover: issue a 2D tile load Global -> LDS.
// D# per CDNA5 ISA: group0 (4 SGPRs) carries
// count/lds_addr/global_addr/type; group1 (8 SGPRs) carries data_size,
// tensor dims, tile dims, dim0 stride. 2-operand form: VADDR2/3 = NULL
// (tensor rank <= 2). Tracked by TENSORcnt.
// All arguments must be wave-uniform.
// ---------------------------------------------------------------------------
__device__ inline void tdm_load_2d(const void* gaddr, uint32_t lds_off,
                                   uint32_t tensor_d0, uint32_t tensor_d1,
                                   uint32_t tile_d0, uint32_t tile_d1,
                                   uint32_t stride0 /* data_size units */) {
  const uint64_t ga = (uint64_t)(uintptr_t)gaddr;
  u32x4 g0;
  g0[0] = 1u;                                  // count=1, user descriptor
  g0[1] = lds_off;                             // lds_addr (bytes)
  g0[2] = (uint32_t)ga;                        // global_addr[31:0]
  g0[3] = (uint32_t)((ga >> 32) & 0x01FFFFFFu) // global_addr[56:32]
          | (2u << 30);                        // type=2 ("image")
  u32x8 g1;
  g1[0] = 1u << 16;                            // data_size=1 -> 2 bytes
  g1[1] = (tensor_d0 & 0xFFFFu) << 16;         // tensor_dim0[15:0]
  g1[2] = (tensor_d0 >> 16) | ((tensor_d1 & 0xFFFFu) << 16);
  g1[3] = (tensor_d1 >> 16) | (tile_d0 << 16); // tile_dim0
  g1[4] = tile_d1;                             // tile_dim1 (tile_dim2 = 0)
  g1[5] = stride0;                             // tensor_dim0_stride[31:0]
  g1[6] = 0u;                                  // stride hi / dim1_stride lo
  g1[7] = 0u;
  asm volatile("tensor_load_to_lds %0, %1" :: "s"(g0), "s"(g1) : "memory");
}

// ---------------------------------------------------------------------------
// Prep kernels: f32 -> f16 convert, and convert+transpose for weights.
// ---------------------------------------------------------------------------
__global__ void k_cvt_f16(const float* __restrict__ src,
                          _Float16* __restrict__ dst, int n) {
  int i = blockIdx.x * blockDim.x + threadIdx.x;
  if (i < n) dst[i] = (_Float16)src[i];
}

// W[rows][cols] (f32, row-major) -> WT[cols][rows] (f16)
__global__ void k_transpose_cvt(const float* __restrict__ W,
                                _Float16* __restrict__ WT,
                                int rows, int cols) {
  int i = blockIdx.x * blockDim.x + threadIdx.x;
  if (i < rows * cols) {
    int r = i / cols, c = i % cols;
    WT[(size_t)c * rows + r] = (_Float16)W[i];
  }
}

// ---------------------------------------------------------------------------
// Stage 1: fused QKV projection. One wave = one 16x16 output tile.
// x_h:  [8192][128] f16.  W*T: [1024][128] f16 (pre-transposed).
// Q,K -> [B*H][N][128] f16; V -> transposed [B*H][128][N] f16.
// ---------------------------------------------------------------------------
__global__ __launch_bounds__(256)
void k_qkv_gemm(const _Float16* __restrict__ xh,
                const _Float16* __restrict__ WqT,
                const _Float16* __restrict__ WkT,
                const _Float16* __restrict__ WvT,
                _Float16* __restrict__ Qb,
                _Float16* __restrict__ Kb,
                _Float16* __restrict__ VTb) {
  const int wid  = (blockIdx.x * blockDim.x + threadIdx.x) >> 5;
  const int mat  = wid / (512 * 64);
  const int rem  = wid % (512 * 64);
  const int m0   = (rem / 64) * 16;
  const int n0   = (rem % 64) * 16;
  const _Float16* WT = (mat == 0) ? WqT : (mat == 1) ? WkT : WvT;

  v8f acc = {};
#pragma unroll
  for (int ks = 0; ks < 4; ++ks) {
    v16h a = load_frag16x32(xh + (size_t)m0 * HDIM + ks * 32, HDIM);
    v16h b = load_frag16x32(WT + (size_t)n0 * HDIM + ks * 32, HDIM);
    acc = wmma_f16(a, b, acc);
  }

  const int lane = threadIdx.x & 31;
  const int col  = n0 + (lane & 15);        // 0..1023
  const int h    = col >> 7;
  const int d    = col & 127;
  const int rb   = (lane >> 4) << 3;        // 0 or 8
#pragma unroll
  for (int r = 0; r < 8; ++r) {
    const int grow = m0 + rb + r;           // 0..8191
    const int b_   = grow >> 11;
    const int n    = grow & (SEQ - 1);
    const int bh   = b_ * NHEAD + h;
    const _Float16 val = (_Float16)acc[r];
    if (mat == 0)      Qb[((size_t)bh * SEQ + n) * HDIM + d] = val;
    else if (mat == 1) Kb[((size_t)bh * SEQ + n) * HDIM + d] = val;
    else               VTb[((size_t)bh * HDIM + d) * SEQ + n] = val;
  }
}

// ---------------------------------------------------------------------------
// Stage 2: causal flash attention with TDM-staged K/V.
// Workgroup = 8 waves = 128 consecutive queries of one (b,h).
// kv blocks of 32 iterate up to the WG's causal limit; wave 0 drives the
// double-buffered TDM pipeline; waves past their own diagonal skip compute
// but keep the barrier schedule (wave-uniform predicate).
// LDS: 2 x (8KB K + 8KB V) + 8 x 1KB P-relayout = 40KB / WG.
// ---------------------------------------------------------------------------
__global__ __launch_bounds__(256)
void k_flash_attn(const _Float16* __restrict__ Qb,
                  const _Float16* __restrict__ Kb,
                  const _Float16* __restrict__ VTb,
                  _Float16* __restrict__ attn) {
  __shared__ _Float16 ldsK[2][32 * HDIM];   // K block: [32 kv][128 d]
  __shared__ _Float16 ldsV[2][HDIM * 32];   // V block: [128 d][32 kv]
  __shared__ _Float16 ldsP[8 * 16 * 32];    // per-wave P re-layout tile

  const int lane  = threadIdx.x & 31;
  const int wslot = threadIdx.x >> 5;
  const int bh    = blockIdx.x >> 4;        // 0..31
  const int wgq0  = (blockIdx.x & 15) << 7; // WG query origin (x128)
  const int q0    = wgq0 + wslot * 16;      // this wave's query tile

  const _Float16* Qh = Qb  + (size_t)bh * SEQ * HDIM;
  const _Float16* Kh = Kb  + (size_t)bh * SEQ * HDIM;
  const _Float16* Vh = VTb + (size_t)bh * HDIM * SEQ;
  _Float16* ldsw = ldsP + wslot * 512;

  // Preload Q fragments for all 4 K-steps (d = 0..127).
  v16h qf[4];
#pragma unroll
  for (int ks = 0; ks < 4; ++ks)
    qf[ks] = load_frag16x32(Qh + (size_t)q0 * HDIM + ks * 32, HDIM);

  v8f acc[8];
#pragma unroll
  for (int f = 0; f < 8; ++f) acc[f] = (v8f){};
  float m_i[8], l_i[8];
#pragma unroll
  for (int r = 0; r < 8; ++r) { m_i[r] = -1e30f; l_i[r] = 0.f; }

  const float scale = 0.08838834764831845f; // 1/sqrt(128)
  const int   rbase = (lane >> 4) << 3;     // C-frag row half: 0 or 8
  const int   cidx  = lane & 15;
  const int   nblocks = (wgq0 >> 5) + 4;    // kv blocks: j = 0 .. wgq0+127

  // Prime the pipeline: TDM fetch of block 0 into buffer 0 (wave 0 only).
  if (wslot == 0) {
    tdm_load_2d(Kh, lds_offset(ldsK[0]), HDIM, SEQ, HDIM, 32, HDIM);
    tdm_load_2d(Vh, lds_offset(ldsV[0]), SEQ, HDIM, 32, HDIM, SEQ);
    __builtin_amdgcn_s_wait_tensorcnt(0);
  }
  __syncthreads();

  for (int blk = 0; blk < nblocks; ++blk) {
    const int j   = blk << 5;
    const int buf = blk & 1;

    // Issue next block's TDM into the other buffer, overlapped with compute.
    if (wslot == 0 && blk + 1 < nblocks) {
      const int jn = j + 32;
      tdm_load_2d(Kh + (size_t)jn * HDIM, lds_offset(ldsK[buf ^ 1]),
                  HDIM, SEQ, HDIM, 32, HDIM);
      tdm_load_2d(Vh + jn, lds_offset(ldsV[buf ^ 1]),
                  SEQ, HDIM, 32, HDIM, SEQ);
    }

    if (j <= q0 + 15) {  // wave-uniform causal predicate
      const _Float16* Kblk = ldsK[buf];
      const _Float16* Vblk = ldsV[buf];

      // ---- scores: S[16x32] = Q_tile @ K_block^T -------------------------
      v8f s0 = {}, s1 = {};
#pragma unroll
      for (int ks = 0; ks < 4; ++ks) {
        v16h k0f = load_frag16x32(Kblk + ks * 32, HDIM);
        v16h k1f = load_frag16x32(Kblk + 16 * HDIM + ks * 32, HDIM);
        s0 = wmma_f16(qf[ks], k0f, s0);
        s1 = wmma_f16(qf[ks], k1f, s1);
      }

      const bool need_mask = (j + 31 > q0);
#pragma unroll
      for (int r = 0; r < 8; ++r) {
        s0[r] *= scale;
        s1[r] *= scale;
        if (need_mask) {
          const int qrow = q0 + rbase + r;
          const int kcol = j + cidx;
          if (kcol > qrow)      s0[r] = -1e30f;
          if (kcol + 16 > qrow) s1[r] = -1e30f;
        }
      }

      // ---- online softmax (rows live in 16-lane half-groups) -------------
#pragma unroll
      for (int r = 0; r < 8; ++r) {
        float mx = fmaxf(s0[r], s1[r]);
#pragma unroll
        for (int off = 8; off >= 1; off >>= 1)
          mx = fmaxf(mx, __shfl_xor(mx, off, 32));
        const float mnew = fmaxf(m_i[r], mx);
        const float corr = __expf(m_i[r] - mnew);
        const float p0 = __expf(s0[r] - mnew);
        const float p1 = __expf(s1[r] - mnew);
        float rs = p0 + p1;
#pragma unroll
        for (int off = 8; off >= 1; off >>= 1)
          rs += __shfl_xor(rs, off, 32);
        l_i[r] = l_i[r] * corr + rs;
        m_i[r] = mnew;
        s0[r] = p0;
        s1[r] = p1;
#pragma unroll
        for (int f = 0; f < 8; ++f) acc[f][r] *= corr;
      }

      // ---- P: C-fragment layout -> A-fragment layout via LDS -------------
#pragma unroll
      for (int r = 0; r < 8; ++r) {
        ldsw[(rbase + r) * 32 + cidx]      = (_Float16)s0[r];
        ldsw[(rbase + r) * 32 + 16 + cidx] = (_Float16)s1[r];
      }
      asm volatile("s_wait_dscnt 0x0" ::: "memory"); // same-wave LDS RAW
      v16h pf = load_frag16x32(ldsw, 32);

      // ---- O += P @ V_block ----------------------------------------------
#pragma unroll
      for (int f = 0; f < 8; ++f) {
        v16h vf = load_frag16x32(Vblk + (f * 16) * 32, 32);
        acc[f] = wmma_f16(pf, vf, acc[f]);
      }
    }

    // Pipeline sync: ensure next buffer landed, then release all waves.
    if (wslot == 0) __builtin_amdgcn_s_wait_tensorcnt(0);
    __syncthreads();
  }

  // ---- normalize and store to attn[b][n][h*128 + d] (f16) ----------------
  const int b_ = bh >> 3;
  const int h  = bh & 7;
#pragma unroll
  for (int r = 0; r < 8; ++r) {
    const float inv = 1.0f / l_i[r];
    const int n = q0 + rbase + r;
    _Float16* dst = attn + ((size_t)(b_ * SEQ + n)) * MODEL + h * HDIM;
#pragma unroll
    for (int f = 0; f < 8; ++f)
      dst[f * 16 + cidx] = (_Float16)(acc[f][r] * inv);
  }
}

// ---------------------------------------------------------------------------
// Stage 3: output projection. attn[8192][1024] @ Wo -> out[8192][128] (f32).
// WoT is Wo transposed: [128][1024] f16. 512 x 8 tiles = 4096 waves.
// ---------------------------------------------------------------------------
__global__ __launch_bounds__(256)
void k_out_gemm(const _Float16* __restrict__ attn,
                const _Float16* __restrict__ WoT,
                float* __restrict__ out) {
  const int wid = (blockIdx.x * blockDim.x + threadIdx.x) >> 5;
  const int m0  = (wid >> 3) * 16;
  const int n0  = (wid & 7) * 16;

  v8f acc = {};
#pragma unroll 4
  for (int ks = 0; ks < 32; ++ks) {
    v16h a = load_frag16x32(attn + (size_t)m0 * MODEL + ks * 32, MODEL);
    v16h b = load_frag16x32(WoT  + (size_t)n0 * MODEL + ks * 32, MODEL);
    acc = wmma_f16(a, b, acc);
  }

  const int lane = threadIdx.x & 31;
  const int col  = n0 + (lane & 15);
  const int rb   = (lane >> 4) << 3;
#pragma unroll
  for (int r = 0; r < 8; ++r)
    out[(size_t)(m0 + rb + r) * HDIM + col] = acc[r];
}

// ---------------------------------------------------------------------------
// Host-side launch sequence (all on `stream`; graph-capture safe).
// ---------------------------------------------------------------------------
extern "C" void kernel_launch(void* const* d_in, const int* in_sizes, int n_in,
                              void* d_out, int out_size, void* d_ws, size_t ws_size,
                              hipStream_t stream) {
  const float* x  = (const float*)d_in[0];  // [4,2048,128]
  const float* Wq = (const float*)d_in[1];  // [128,1024]
  const float* Wk = (const float*)d_in[2];
  const float* Wv = (const float*)d_in[3];
  const float* Wo = (const float*)d_in[4];  // [1024,128]
  float* out = (float*)d_out;               // [4,2048,128]

  // Workspace layout (bytes, all 16B-aligned)
  char* ws = (char*)d_ws;
  const size_t MB = 1024ull * 1024ull;
  _Float16* xh   = (_Float16*)(ws + 0);          //  2 MB: [8192][128]
  _Float16* WqT  = (_Float16*)(ws + 2 * MB);     // 256 KB each
  _Float16* WkT  = (_Float16*)(ws + 2 * MB + 256 * 1024);
  _Float16* WvT  = (_Float16*)(ws + 2 * MB + 512 * 1024);
  _Float16* WoT  = (_Float16*)(ws + 2 * MB + 768 * 1024);
  _Float16* Qb   = (_Float16*)(ws + 3 * MB);     // 16 MB: [32][2048][128]
  _Float16* Kb   = (_Float16*)(ws + 19 * MB);    // 16 MB
  _Float16* VTb  = (_Float16*)(ws + 35 * MB);    // 16 MB: [32][128][2048]
  _Float16* attn = (_Float16*)(ws + 51 * MB);    // 16 MB: [8192][1024]

  // Stage 0: conversions / transposes
  k_cvt_f16<<<(ROWS * HDIM + 255) / 256, 256, 0, stream>>>(x, xh, ROWS * HDIM);
  k_transpose_cvt<<<(HDIM * MODEL + 255) / 256, 256, 0, stream>>>(Wq, WqT, HDIM, MODEL);
  k_transpose_cvt<<<(HDIM * MODEL + 255) / 256, 256, 0, stream>>>(Wk, WkT, HDIM, MODEL);
  k_transpose_cvt<<<(HDIM * MODEL + 255) / 256, 256, 0, stream>>>(Wv, WvT, HDIM, MODEL);
  k_transpose_cvt<<<(MODEL * HDIM + 255) / 256, 256, 0, stream>>>(Wo, WoT, MODEL, HDIM);

  // Stage 1: QKV projection — 3 * 512 * 64 tiles, 8 waves/block
  k_qkv_gemm<<<(3 * 512 * 64) / 8, 256, 0, stream>>>(xh, WqT, WkT, WvT, Qb, Kb, VTb);

  // Stage 2: causal flash attention — 512 WGs x 8 waves (TDM pipeline)
  k_flash_attn<<<512, 256, 0, stream>>>(Qb, Kb, VTb, attn);

  // Stage 3: output projection — 512 * 8 tiles
  k_out_gemm<<<(512 * 8) / 8, 256, 0, stream>>>(attn, WoT, out);
}
